// PAM_12833362280786
// MI455X (gfx1250) — matrix operand order repaired and checked by reading, not compiled, not run on hardware
//
#include <hip/hip_runtime.h>
#include <hip/hip_bf16.h>

// ---------------------------------------------------------------------------
// PAM attention for MI455X (gfx1250, wave32, WMMA).
//   x:(4,512,64,64) f32; Wb,Wc:(64,512); Wd:(512,512); gamma scalar.
//   out = gamma * attention(x) + x, all matmuls via v_wmma_f32_16x16x32_bf16.
// Workspace layout (bytes):
//   [0      , 640*512*2)  Wcat bf16 (rows: 0-63 Wb, 64-127 Wc, 128-639 Wd)
//   [1 MiB  , +2 MiB)     Q  (B, N, 64)  bf16   (queries, row-major over m)
//   [3 MiB  , +2 MiB)     Kt (B, N, 64)  bf16   (keys^T,  row-major over m)
//   [5 MiB  , +16 MiB)    Vt (B, 512, N) bf16   (values,  row-major over n)
// Total 21 MiB.
// ---------------------------------------------------------------------------

typedef __bf16 bf16_t;
typedef __attribute__((ext_vector_type(16))) __bf16 v16bf;
typedef __attribute__((ext_vector_type(8)))  __bf16 v8bf;
typedef __attribute__((ext_vector_type(8)))  float  v8f;

#define WMMA_BF16(a, b, c) \
  __builtin_amdgcn_wmma_f32_16x16x32_bf16(false, (a), false, (b), (short)0, (c), false, false)

static constexpr int CB   = 512;   // channels
static constexpr int MID  = 64;    // query/key dim
static constexpr int NB   = 4096;  // H*W
static constexpr int BB   = 4;     // batch

// A-fragment (16x32, bf16) from row-major src (stride ld in elements).
// Lane 0-15: row=row0+l16, K 0..15; lane 16-31: same row, K 16..31.
__device__ __forceinline__ v16bf load_afrag(const bf16_t* src, int ld, int row0,
                                            int kbase, int l16, int hi) {
  const bf16_t* p = src + (size_t)(row0 + l16) * ld + kbase + hi * 16;
  v16bf r;
  *(v8bf*)&r       = *(const v8bf*)p;
  *((v8bf*)&r + 1) = *(const v8bf*)(p + 8);
  return r;
}

// B-fragment (32x16, bf16) from (col, k) row-major src (stride ld in elements).
// Lane 0-15: col=col0+l16, K 0..15; lane 16-31: same col, K 16..31.
__device__ __forceinline__ v16bf load_bfrag(const bf16_t* src, int ld, int col0,
                                            int kbase, int l16, int hi) {
  const bf16_t* p = src + (size_t)(col0 + l16) * ld + kbase + hi * 16;
  v16bf r;
  *(v8bf*)&r       = *(const v8bf*)p;
  *((v8bf*)&r + 1) = *(const v8bf*)(p + 8);
  return r;
}

// ---------------------------------------------------------------------------
// Kernel 1: build concatenated bf16 weight matrix Wcat (640 x 512).
// ---------------------------------------------------------------------------
__global__ void pam_prep_w(const float* __restrict__ Wb, const float* __restrict__ Wc,
                           const float* __restrict__ Wd, bf16_t* __restrict__ wcat) {
  int i = blockIdx.x * 256 + threadIdx.x;
  if (i >= 640 * CB) return;
  int r = i >> 9, c = i & 511;
  float v;
  if (r < 64)        v = Wb[r * CB + c];
  else if (r < 128)  v = Wc[(r - 64) * CB + c];
  else               v = Wd[(size_t)(r - 128) * CB + c];
  wcat[i] = (bf16_t)v;
}

// ---------------------------------------------------------------------------
// Kernel 2: fused projections.  For each batch: Wcat(640x512) x x(512xN).
// Rows 0-63 -> Q (N,64), rows 64-127 -> Kt (N,64), rows 128-639 -> Vt (512,N).
// Grid: (N/64, 640/64, B), block 128 (4 waves); wave w owns 16 rows x 64 cols.
// ---------------------------------------------------------------------------
__global__ __launch_bounds__(128) void pam_proj(
    const float* __restrict__ x, const bf16_t* __restrict__ wcat,
    bf16_t* __restrict__ q, bf16_t* __restrict__ kt, bf16_t* __restrict__ vt) {
  __shared__ bf16_t ldsx[64 * 32];  // [n][c] transposed x chunk, bf16

  const int tid  = threadIdx.x;
  const int wave = tid >> 5;
  const int lane = tid & 31;
  const int l16  = lane & 15;
  const int hi   = lane >> 4;

  const int n0 = blockIdx.x * 64;
  const int r0 = blockIdx.y * 64;
  const int b  = blockIdx.z;
  const float* xb = x + (size_t)b * CB * NB;

  v8f acc[4];
#pragma unroll
  for (int t = 0; t < 4; ++t)
#pragma unroll
    for (int r = 0; r < 8; ++r) acc[t][r] = 0.0f;

  for (int c0 = 0; c0 < CB; c0 += 32) {
    __syncthreads();
    // Stage x[c0..c0+31][n0..n0+63] -> ldsx[n][c] (bf16, transposed).
#pragma unroll
    for (int i = tid; i < 32 * 64; i += 128) {
      int c = i >> 6, n = i & 63;
      ldsx[n * 32 + c] = (bf16_t)xb[(size_t)(c0 + c) * NB + n0 + n];
    }
    __syncthreads();

    v16bf aw = load_afrag(wcat, CB, r0 + wave * 16, c0, l16, hi);
#pragma unroll
    for (int ct = 0; ct < 4; ++ct) {
      const bf16_t* p = &ldsx[(ct * 16 + l16) * 32 + hi * 16];
      v16bf bx;
      *(v8bf*)&bx       = *(const v8bf*)p;
      *((v8bf*)&bx + 1) = *(const v8bf*)(p + 8);
      acc[ct] = WMMA_BF16(aw, bx, acc[ct]);
    }
  }

  const int Rbase = r0 + wave * 16 + hi * 8;
#pragma unroll
  for (int ct = 0; ct < 4; ++ct) {
    int n = n0 + ct * 16 + l16;
#pragma unroll
    for (int r = 0; r < 8; ++r) {
      int R = Rbase + r;
      bf16_t v = (bf16_t)acc[ct][r];
      if (R < 64)        q[((size_t)b * NB + n) * MID + R] = v;
      else if (R < 128)  kt[((size_t)b * NB + n) * MID + (R - 64)] = v;
      else               vt[((size_t)b * CB + (R - 128)) * (size_t)NB + n] = v;
    }
  }
}

// ---------------------------------------------------------------------------
// Kernel 3: flash attention + epilogue (gamma*out + x).
// Grid (N/32, B), block 256 (8 waves). Wave = (row-tile rt in {0,1},
// channel group cg in {0..3} covering 128 output channels = 8 col-tiles).
// Streams over 64-key blocks with online softmax; O accum stays in VGPRs.
// ---------------------------------------------------------------------------
__global__ __launch_bounds__(256) void pam_attn(
    const float* __restrict__ x, const bf16_t* __restrict__ q,
    const bf16_t* __restrict__ kt, const bf16_t* __restrict__ vt,
    const float* __restrict__ gamma, float* __restrict__ out) {
  __shared__ bf16_t Pbuf[8][16 * 64];  // per-wave P tile (16 queries x 64 keys)

  const int tid  = threadIdx.x;
  const int wave = tid >> 5;
  const int lane = tid & 31;
  const int l16  = lane & 15;
  const int hi   = lane >> 4;

  const int b  = blockIdx.y;
  const int qb = blockIdx.x * 32;
  const int rt = wave >> 2;  // row tile (16 queries)
  const int cg = wave & 3;   // 128-channel group

  const bf16_t* qB = q  + (size_t)b * NB * MID;
  const bf16_t* kB = kt + (size_t)b * NB * MID;
  const bf16_t* vB = vt + (size_t)b * CB * NB;

  const int row0 = qb + rt * 16;
  // Q fragments for this wave's 16 query rows (MID=64 -> 2 k-steps of 32).
  v16bf aq0 = load_afrag(qB, MID, row0, 0, l16, hi);
  v16bf aq1 = load_afrag(qB, MID, row0, 32, l16, hi);

  v8f oacc[8];
#pragma unroll
  for (int t = 0; t < 8; ++t)
#pragma unroll
    for (int r = 0; r < 8; ++r) oacc[t][r] = 0.0f;

  float mrow[8], lrow[8];
#pragma unroll
  for (int r = 0; r < 8; ++r) { mrow[r] = -1e30f; lrow[r] = 0.0f; }

  bf16_t* myP = Pbuf[wave];

  for (int kb = 0; kb < NB; kb += 64) {
    // ---- S = Q * K^T for 4 key tiles of 16 ----
    v8f S[4];
#pragma unroll
    for (int j = 0; j < 4; ++j) {
      v8f s;
#pragma unroll
      for (int r = 0; r < 8; ++r) s[r] = 0.0f;
      v16bf bk0 = load_bfrag(kB, MID, kb + j * 16, 0, l16, hi);
      s = WMMA_BF16(aq0, bk0, s);
      v16bf bk1 = load_bfrag(kB, MID, kb + j * 16, 32, l16, hi);
      s = WMMA_BF16(aq1, bk1, s);
      S[j] = s;
    }

    // ---- online softmax (rows live per half-wave: lanes 0-15 rows r, 16-31 rows 8+r) ----
#pragma unroll
    for (int r = 0; r < 8; ++r) {
      float mx = fmaxf(fmaxf(S[0][r], S[1][r]), fmaxf(S[2][r], S[3][r]));
      mx = fmaxf(mx, __shfl_xor(mx, 1, 16));
      mx = fmaxf(mx, __shfl_xor(mx, 2, 16));
      mx = fmaxf(mx, __shfl_xor(mx, 4, 16));
      mx = fmaxf(mx, __shfl_xor(mx, 8, 16));
      float mnew  = fmaxf(mrow[r], mx);
      float scale = __expf(mrow[r] - mnew);
      mrow[r] = mnew;
      lrow[r] *= scale;
#pragma unroll
      for (int t = 0; t < 8; ++t) oacc[t][r] *= scale;
      float ps = 0.0f;
#pragma unroll
      for (int j = 0; j < 4; ++j) {
        float p = __expf(S[j][r] - mnew);
        ps += p;
        myP[(hi * 8 + r) * 64 + j * 16 + l16] = (bf16_t)p;
      }
      ps += __shfl_xor(ps, 1, 16);
      ps += __shfl_xor(ps, 2, 16);
      ps += __shfl_xor(ps, 4, 16);
      ps += __shfl_xor(ps, 8, 16);
      lrow[r] += ps;
    }
    // Same-wave LDS ops are in-order: P stores complete before the loads below.

    // ---- O += P * V (2 k-steps of 32 keys, 8 channel tiles) ----
#pragma unroll
    for (int ks = 0; ks < 2; ++ks) {
      const bf16_t* pp = myP + l16 * 64 + ks * 32 + hi * 16;
      v16bf ap;
      *(v8bf*)&ap       = *(const v8bf*)pp;
      *((v8bf*)&ap + 1) = *(const v8bf*)(pp + 8);
#pragma unroll
      for (int t = 0; t < 8; ++t) {
        v16bf bv = load_bfrag(vB, NB, cg * 128 + t * 16, kb + ks * 32, l16, hi);
        oacc[t] = WMMA_BF16(ap, bv, oacc[t]);
      }
    }
  }

  // ---- epilogue: out = gamma * (O / l) + x ----
  const float g = gamma[0];
  const float* xb = x   + (size_t)b * CB * NB;
  float*       ob = out + (size_t)b * CB * NB;
#pragma unroll
  for (int r = 0; r < 8; ++r) {
    float inv = 1.0f / lrow[r];
    int n = row0 + hi * 8 + r;
#pragma unroll
    for (int t = 0; t < 8; ++t) {
      int c = cg * 128 + t * 16 + l16;
      size_t idx = (size_t)c * NB + n;
      ob[idx] = g * (oacc[t][r] * inv) + xb[idx];
    }
  }
}

// ---------------------------------------------------------------------------
extern "C" void kernel_launch(void* const* d_in, const int* in_sizes, int n_in,
                              void* d_out, int out_size, void* d_ws, size_t ws_size,
                              hipStream_t stream) {
  const float* x     = (const float*)d_in[0];
  const float* Wb    = (const float*)d_in[1];
  const float* Wc    = (const float*)d_in[2];
  const float* Wd    = (const float*)d_in[3];
  const float* gamma = (const float*)d_in[4];
  float* out = (float*)d_out;

  char* ws = (char*)d_ws;
  bf16_t* wcat = (bf16_t*)(ws);
  bf16_t* q    = (bf16_t*)(ws + (1u << 20));
  bf16_t* kt   = (bf16_t*)(ws + (3u << 20));
  bf16_t* vt   = (bf16_t*)(ws + (5u << 20));

  pam_prep_w<<<(640 * CB + 255) / 256, 256, 0, stream>>>(Wb, Wc, Wd, wcat);
  pam_proj<<<dim3(NB / 64, 640 / 64, BB), 128, 0, stream>>>(x, wcat, q, kt, vt);
  pam_attn<<<dim3(NB / 32, BB), 256, 0, stream>>>(x, q, kt, vt, gamma, out);
}
